// ChunkedLocalAttention_16449724745442
// MI455X (gfx1250) — matrix operand order repaired and checked by reading, not compiled, not run on hardware
//
#include <hip/hip_runtime.h>
#include <hip/hip_bf16.h>
#include <math.h>

// ---------------- CDNA5 WMMA types ----------------
typedef __attribute__((ext_vector_type(16))) __bf16 v16bf;
typedef __attribute__((ext_vector_type(8)))  float  v8f;

union Frag {
    v16bf v;
    uint4 q[2];
};

__device__ __forceinline__ int imind(int a, int b) { return a < b ? a : b; }
__device__ __forceinline__ int imaxd(int a, int b) { return a > b ? a : b; }

__device__ __forceinline__ v8f wmma_bf16(const Frag& a, const Frag& b, v8f c) {
    // D = A(16x32 bf16) * B(32x16 bf16) + C(16x16 f32)
    return __builtin_amdgcn_wmma_f32_16x16x32_bf16(
        /*neg_a=*/false, a.v, /*neg_b=*/false, b.v,
        /*c_mod=*/(short)0, c, /*reuse_a=*/false, /*reuse_b=*/false);
}

// Problem constants
#define BATCH   8
#define TLEN    4096
#define DMODEL  512
#define NHEADS  8
#define HDIM    64
#define WCHUNK  128
#define NCHUNKS 32          // U = T / W
#define CTX     287
#define CPAD    288
#define FSPAN   160
#define VSTR    296         // padded LDS stride (elems) for probs / v^T
#define QSCALE_CONST 0.18033688011112042f   // 64^-0.5 / ln(2)

// ---------------------------------------------------------------------------
// Prep kernels
// ---------------------------------------------------------------------------
__global__ void cla_transpose_w(const float* __restrict__ in, __bf16* __restrict__ out) {
    // in: [512][512] f32 (k-major), out: [512][512] bf16 = in^T (n-major)
    int idx = blockIdx.x * 256 + threadIdx.x;       // 262144 total
    int j = idx >> 9, k = idx & 511;
    out[idx] = (__bf16)in[(size_t)k * 512 + j];
}

__global__ void cla_timing_signal(float* __restrict__ ts) {
    // ts: [160][512] f32 sinusoid table, pos[f] = 127 - f
    int idx = blockIdx.x * 256 + threadIdx.x;       // 81920 total
    int f = idx >> 9, j = idx & 511;
    float pos = 127.0f - (float)f;
    int i = j & 255;
    float log_inc = 9.210340371976184f / 255.0f;    // log(10000)/(num_ts-1)
    float inv = __expf(-(float)i * log_inc);
    float s = pos * inv;
    ts[idx] = (j >= 256) ? __cosf(s) : __sinf(s);
}

__global__ void cla_qscale(const float* __restrict__ pds, float* __restrict__ qs) {
    int h = threadIdx.x;                            // 64 threads
    float sp = log1pf(__expf(pds[h]));              // softplus
    qs[h] = QSCALE_CONST * sp;
}

// ---------------------------------------------------------------------------
// WMMA GEMM: out[M x 512] = A[M x 512](f32) @ B (given as BT[512 x 512] bf16)
// MODE 0: bf16 row-major out  (k, v)           -- M assumed multiple of 128
// MODE 1: bf16 row-major out scaled by qscale  (q) -- M multiple of 128
// MODE 2: bf16 out at [(col>>6)*160 + row]*64 + (col&63)  (sin_emb [n][f][h])
// ---------------------------------------------------------------------------
template <int MODE>
__global__ __launch_bounds__(256) void cla_gemm_bf16(
    const float* __restrict__ A, const __bf16* __restrict__ BT,
    __bf16* __restrict__ out, int M, const float* __restrict__ qscale)
{
    const int lane = threadIdx.x & 31;
    const int wave = threadIdx.x >> 5;
    const int hi = lane >> 4;       // half-lane select
    const int ln = lane & 15;

    const int row0  = blockIdx.x * 128 + wave * 16;
    const int nbase = blockIdx.y * 64;

    int arow = imind(row0 + ln, M - 1);
    const float* aptr = A + (size_t)arow * 512;

    // q-scale per output column, hoisted out of the epilogue:
    // nbase is a multiple of 64 -> col & 63 == j*16 + ln
    float scl[4] = {1.f, 1.f, 1.f, 1.f};
    if (MODE == 1) {
#pragma unroll
        for (int j = 0; j < 4; ++j) scl[j] = qscale[j * 16 + ln];
    }

    v8f acc[4] = {};

    for (int kb = 0; kb < 16; ++kb) {
        // ---- A fragment: row = ln, K in {k0..k0+7, k0+16..k0+23}
        int k0 = kb * 32 + 8 * hi;
        union { float4 f4[4]; float f[16]; } ua;
        ua.f4[0] = *(const float4*)(aptr + k0);
        ua.f4[1] = *(const float4*)(aptr + k0 + 4);
        ua.f4[2] = *(const float4*)(aptr + k0 + 16);
        ua.f4[3] = *(const float4*)(aptr + k0 + 20);
        if (kb < 15) __builtin_prefetch(aptr + k0 + 32, 0, 3);   // global_prefetch_b8
        Frag af;
#pragma unroll
        for (int i = 0; i < 16; ++i) af.v[i] = (__bf16)ua.f[i];

        // ---- 4 B fragments: column n = nbase+j*16+ln, K in {k0..k0+7, k0+16..k0+23}
#pragma unroll
        for (int j = 0; j < 4; ++j) {
            const __bf16* bptr = BT + (size_t)(nbase + j * 16 + ln) * 512 + kb * 32 + 8 * hi;
            Frag bf_;
            bf_.q[0] = *(const uint4*)(bptr);
            bf_.q[1] = *(const uint4*)(bptr + 16);
            acc[j] = wmma_bf16(af, bf_, acc[j]);
        }
    }

    // Epilogue: C/D layout row = row0 + r + 8*hi, col = nbase + j*16 + ln
#pragma unroll
    for (int j = 0; j < 4; ++j) {
        const int col = nbase + j * 16 + ln;
#pragma unroll
        for (int r = 0; r < 8; ++r) {
            const int row = row0 + r + 8 * hi;
            if (MODE == 2 && row >= M) continue;
            float v = acc[j][r];
            if (MODE == 1) v *= scl[j];
            __bf16 bv = (__bf16)v;
            if (MODE == 2) {
                int n = col >> 6, h = col & 63;
                out[((size_t)n * FSPAN + row) * 64 + h] = bv;
            } else {
                out[(size_t)row * 512 + col] = bv;
            }
        }
    }
}

// ---------------------------------------------------------------------------
// Attention kernel: one workgroup per (b*u, head, half) handling 64 query rows.
// 8 waves: wave = rt*2 + ch  (rt = 16-row tile 0..3, ch = column half)
// Dynamic LDS:
//   [0, 73728)      : f32 ac scores  [64][288]   -> later reused as bf16 v^T [64][VSTR]
//   [73728, 114688) : f32 bd scores  [64][160]   -> later reused as bf16 probs [64][VSTR]
// ---------------------------------------------------------------------------
__global__ __launch_bounds__(256) void cla_attn(
    const __bf16* __restrict__ qb, const __bf16* __restrict__ kb,
    const __bf16* __restrict__ vb, const __bf16* __restrict__ sinb,
    const unsigned char* __restrict__ mask, float* __restrict__ out)
{
    extern __shared__ char smem[];
    float*  ac    = (float*)smem;                      // [64][288]
    float*  bd    = (float*)(smem + 64 * CPAD * 4);    // [64][160]
    __bf16* probs = (__bf16*)bd;                       // [64][VSTR] bf16 (36.9KB <= 40KB)
    __bf16* vt    = (__bf16*)smem;                     // [64][VSTR] bf16 (37.9KB <= 72KB)

    const int tid  = threadIdx.x;
    const int lane = tid & 31;
    const int wave = tid >> 5;
    const int hi   = lane >> 4;
    const int ln   = lane & 15;

    const int bu   = blockIdx.x;            // 0..255
    const int b    = bu >> 5;
    const int u    = bu & 31;
    const int n    = blockIdx.y;            // head
    const int half = blockIdx.z;            // 0/1 -> rows [0,64) or [64,128)
    const int wbase = half * 64;
    const int t0    = u * WCHUNK;

    const int rt = wave >> 1;               // row tile within the 64 rows
    const int ch = wave & 1;                // column half

    // ---- Q fragments (16 rows x 64 K => 2 fragments) for this wave's row tile
    const int tq = t0 + wbase + rt * 16 + ln;
    const __bf16* qp = qb + ((size_t)(b * TLEN + tq)) * 512 + n * 64;
    Frag qf0, qf1;
    {
        int k0 = 16 * hi;
        qf0.q[0] = *(const uint4*)(qp + k0);
        qf0.q[1] = *(const uint4*)(qp + k0 + 8);
        qf1.q[0] = *(const uint4*)(qp + 32 + k0);
        qf1.q[1] = *(const uint4*)(qp + 32 + k0 + 8);
    }

    // ---- AC = q @ k^T  : 9 column tiles per wave (288 cols / 2 halves / 16)
    for (int ct = 0; ct < 9; ++ct) {
        int c0 = (ch * 9 + ct) * 16;
        int tk = t0 - 127 + c0 + ln;                      // context col -> time
        int tkc = imind(imaxd(tk, 0), TLEN - 1);          // clamp (masked later)
        const __bf16* kp = kb + ((size_t)(b * TLEN + tkc)) * 512 + n * 64 + 16 * hi;
        Frag b0, b1;
        b0.q[0] = *(const uint4*)(kp);       b0.q[1] = *(const uint4*)(kp + 8);
        b1.q[0] = *(const uint4*)(kp + 32);  b1.q[1] = *(const uint4*)(kp + 40);
        v8f a = {};
        a = wmma_bf16(qf0, b0, a);
        a = wmma_bf16(qf1, b1, a);
#pragma unroll
        for (int r = 0; r < 8; ++r)
            ac[(rt * 16 + r + 8 * hi) * CPAD + c0 + ln] = a[r];
    }

    // ---- BD = q @ sin_emb^T : 5 column tiles per wave (160 / 2 / 16)
    for (int ct = 0; ct < 5; ++ct) {
        int f0 = (ch * 5 + ct) * 16;
        const __bf16* sp = sinb + ((size_t)n * FSPAN + f0 + ln) * 64 + 16 * hi;
        Frag b0, b1;
        b0.q[0] = *(const uint4*)(sp);       b0.q[1] = *(const uint4*)(sp + 8);
        b1.q[0] = *(const uint4*)(sp + 32);  b1.q[1] = *(const uint4*)(sp + 40);
        v8f a = {};
        a = wmma_bf16(qf0, b0, a);
        a = wmma_bf16(qf1, b1, a);
#pragma unroll
        for (int r = 0; r < 8; ++r)
            bd[(rt * 16 + r + 8 * hi) * FSPAN + f0 + ln] = a[r];
    }
    __syncthreads();

    // ---- combine: rel-shift (bd[w, c-w]), tanh soft-cap, causal + range mask
    for (int idx = tid; idx < 64 * CPAD; idx += 256) {
        int wl = idx / CPAD, c = idx - wl * CPAD;
        int w = wbase + wl;
        int f = c - w;                          // relative-shift trick
        float v = ac[idx];
        if (f >= 0 && f < FSPAN) v += bd[wl * FSPAN + f];
        v = 50.0f * tanhf(v * 0.02f);
        int tk = t0 + c - 127;
        bool valid = (f >= 0) && (f < FSPAN) && (c < CTX) && (tk >= 0) && (tk < TLEN);
        if (valid) valid = (mask[b * TLEN + tk] == 0);
        ac[idx] = valid ? v : -3.0e38f;
    }
    __syncthreads();

    // ---- softmax: 4 lanes per row, shuffle reductions; write bf16 probs
    {
        int wl = tid >> 2, qd = tid & 3;
        const float* rowp = ac + wl * CPAD;
        float m = -3.4e38f;
        for (int c = qd * 72; c < qd * 72 + 72; ++c) m = fmaxf(m, rowp[c]);
        m = fmaxf(m, __shfl_xor(m, 1));
        m = fmaxf(m, __shfl_xor(m, 2));
        float s = 0.0f;
        for (int c = qd * 72; c < qd * 72 + 72; ++c) s += __expf(rowp[c] - m);
        s += __shfl_xor(s, 1);
        s += __shfl_xor(s, 2);
        float inv = 1.0f / s;
        for (int c = qd * 72; c < qd * 72 + 72; ++c)
            probs[wl * VSTR + c] = (__bf16)(__expf(rowp[c] - m) * inv);
    }
    __syncthreads();

    // ---- stage v^T into LDS: vt[h][c] (overwrites ac region)
    for (int idx = tid; idx < CPAD * 8; idx += 256) {
        int c = idx >> 3, hb = idx & 7;
        int tk = t0 + c - 127;
        int tkc = imind(imaxd(tk, 0), TLEN - 1);
        uint4 d = *(const uint4*)(vb + ((size_t)(b * TLEN + tkc)) * 512 + n * 64 + hb * 8);
        const __bf16* dv = (const __bf16*)&d;
#pragma unroll
        for (int j = 0; j < 8; ++j)
            vt[(hb * 8 + j) * VSTR + c] = dv[j];
    }
    __syncthreads();

    // ---- PV: out[w, h] = probs[w, :] @ v[:, h]; 2 output tiles per wave
    {
        const int rt2 = wave >> 1;
        const int chh = wave & 1;
        v8f o0 = {}, o1 = {};
        for (int kc = 0; kc < CPAD; kc += 32) {
            Frag pa;
            const __bf16* pp = probs + (size_t)(rt2 * 16 + ln) * VSTR + kc + 16 * hi;
            pa.q[0] = *(const uint4*)(pp);
            pa.q[1] = *(const uint4*)(pp + 8);
            Frag bv0, bv1;
            const __bf16* vp0 = vt + (size_t)(chh * 32 + ln) * VSTR + kc + 16 * hi;
            bv0.q[0] = *(const uint4*)(vp0);  bv0.q[1] = *(const uint4*)(vp0 + 8);
            const __bf16* vp1 = vt + (size_t)(chh * 32 + 16 + ln) * VSTR + kc + 16 * hi;
            bv1.q[0] = *(const uint4*)(vp1);  bv1.q[1] = *(const uint4*)(vp1 + 8);
            o0 = wmma_bf16(pa, bv0, o0);
            o1 = wmma_bf16(pa, bv1, o1);
        }
#pragma unroll
        for (int r = 0; r < 8; ++r) {
            int t = t0 + wbase + rt2 * 16 + r + 8 * hi;
            float* op = out + (((size_t)(b * TLEN + t)) * NHEADS + n) * 64;
            op[chh * 32 + ln]      = o0[r];
            op[chh * 32 + 16 + ln] = o1[r];
        }
    }
}

// ---------------------------------------------------------------------------
// Host launcher
// ---------------------------------------------------------------------------
extern "C" void kernel_launch(void* const* d_in, const int* in_sizes, int n_in,
                              void* d_out, int out_size, void* d_ws, size_t ws_size,
                              hipStream_t stream) {
    const float*         x    = (const float*)d_in[0];
    const unsigned char* mask = (const unsigned char*)d_in[1];
    const float*         wq   = (const float*)d_in[2];
    const float*         wk   = (const float*)d_in[3];
    const float*         wv   = (const float*)d_in[4];
    const float*         wpos = (const float*)d_in[5];
    const float*         pds  = (const float*)d_in[6];
    float* out = (float*)d_out;

    char* ws = (char*)d_ws;
    __bf16* q_bf   = (__bf16*)(ws);                       // 32 MB
    __bf16* k_bf   = (__bf16*)(ws + 33554432);            // 32 MB
    __bf16* v_bf   = (__bf16*)(ws + 67108864);            // 32 MB
    __bf16* wqT    = (__bf16*)(ws + 100663296);           // 4 x 512KB
    __bf16* wkT    = wqT + 262144;
    __bf16* wvT    = wkT + 262144;
    __bf16* wpT    = wvT + 262144;
    float*  ts     = (float*)(ws + 102760448);            // 160x512 f32
    __bf16* sin_bf = (__bf16*)(ws + 103088128);           // [8][160][64]
    float*  qscale = (float*)(ws + 103251968);            // 64 f32

    // ---- prep
    cla_transpose_w<<<1024, 256, 0, stream>>>(wq,   wqT);
    cla_transpose_w<<<1024, 256, 0, stream>>>(wk,   wkT);
    cla_transpose_w<<<1024, 256, 0, stream>>>(wv,   wvT);
    cla_transpose_w<<<1024, 256, 0, stream>>>(wpos, wpT);
    cla_timing_signal<<<320, 256, 0, stream>>>(ts);
    cla_qscale<<<1, 64, 0, stream>>>(pds, qscale);

    // ---- projections (WMMA): M = B*T = 32768, K = 512, N = 512
    dim3 gq(256, 8);
    cla_gemm_bf16<1><<<gq, 256, 0, stream>>>(x, wqT, q_bf, 32768, qscale);
    cla_gemm_bf16<0><<<gq, 256, 0, stream>>>(x, wkT, k_bf, 32768, qscale);
    cla_gemm_bf16<0><<<gq, 256, 0, stream>>>(x, wvT, v_bf, 32768, qscale);
    // sin_emb = timing_signal @ w_pos : M = 160
    dim3 gs(2, 8);
    cla_gemm_bf16<2><<<gs, 256, 0, stream>>>(ts, wpT, sin_bf, 160, qscale);

    // ---- attention
    (void)hipFuncSetAttribute((const void*)cla_attn,
                              hipFuncAttributeMaxDynamicSharedMemorySize, 114688);
    dim3 ga(BATCH * NCHUNKS, NHEADS, 2);
    cla_attn<<<ga, 256, 114688, stream>>>(q_bf, k_bf, v_bf, sin_bf, mask, out);
}
